// Network_7550552507256
// MI455X (gfx1250) — matrix operand order repaired and checked by reading, not compiled, not run on hardware
//
#include <hip/hip_runtime.h>
#include <hip/hip_bf16.h>

// ---------------------------------------------------------------------------
// Types for CDNA5 WMMA (wave32): v_wmma_f32_16x16x32_bf16
// ---------------------------------------------------------------------------
typedef __attribute__((ext_vector_type(8)))  __bf16 bf16x8;
typedef __attribute__((ext_vector_type(16))) __bf16 bf16x16;
typedef __attribute__((ext_vector_type(8)))  float  f32x8;

#define Vdim 10000
#define Edim 512
#define HNdim 1024
#define HHdim 256
#define Ddepth 4
#define Tdim 128
#define Bdim 64

// padded Wout rows so the 128-wide N block tile divides evenly (79 * 128)
#define NPAD 10112
#define BLK_N 128
#define BLK_M 64
#define LDSW 40   // LDS row stride in bf16 elems: 80B = 16B-aligned, 20-bank stride

// ---------------------------------------------------------------------------
// CDNA5 async Global->LDS path (ASYNCcnt). The builtin is declared by this
// toolchain (probe round 3); its b128 form takes typed v4i pointers:
//   (v4i addrspace(1)* src, v4i addrspace(3)* dst, imm offset, imm cpol)
// Guarded: falls back to global_load + ds_store staging if not declared.
// ---------------------------------------------------------------------------
#if defined(__has_builtin)
#if __has_builtin(__builtin_amdgcn_global_load_async_to_lds_b128)
#define HAVE_ASYNC_LDS 1
#endif
#endif
#ifndef HAVE_ASYNC_LDS
#define HAVE_ASYNC_LDS 0
#endif

typedef int v4i_b128 __attribute__((vector_size(16)));
typedef __attribute__((address_space(1))) v4i_b128* gptr_b128;
typedef __attribute__((address_space(3))) v4i_b128* lptr_b128;

__device__ __forceinline__ gptr_b128 as_global_b128(const void* p) {
    return (gptr_b128)(unsigned long long)p;            // 64-bit global address
}
__device__ __forceinline__ lptr_b128 as_lds_b128(void* p) {
    // generic pointers to LDS carry the LDS byte offset in the low 32 bits
    return (lptr_b128)(unsigned int)(unsigned long long)p;
}

// RNE float -> bf16 bits
__device__ __forceinline__ unsigned short f2bf(float f) {
    unsigned int u = __float_as_uint(f);
    unsigned int r = u + 0x7FFFu + ((u >> 16) & 1u);
    return (unsigned short)(r >> 16);
}

// ---------------------------------------------------------------------------
// Fragment loaders matching the CDNA5 wave32 WMMA VGPR layouts (ISA 7.12.2).
// A (16x32, 16-bit, row-major activations [*, lda]):
//   lane l<16 : row m0+l,    K = k0+0..7  (VGPR0-3) and k0+16..23 (VGPR4-7)
//   lane l>=16: row m0+l-16, K = k0+8..15            and k0+24..31
// B (32x16, from weights stored [N, K] row-major):
//   lane l: column n0+(l&15), K same as A: k0+(l<16?0:8)+0..7 and +16..23
// ---------------------------------------------------------------------------
__device__ __forceinline__ bf16x16 load_a_frag(const unsigned short* A, int lda,
                                               int m0, int k0, int lane) {
    const unsigned short* p =
        A + (size_t)(m0 + (lane & 15)) * lda + k0 + ((lane >> 4) << 3);
    union { bf16x16 v; bf16x8 h[2]; } u;
    u.h[0] = *(const bf16x8*)(p);
    u.h[1] = *(const bf16x8*)(p + 16);
    return u.v;
}

__device__ __forceinline__ bf16x16 load_b_frag(const unsigned short* W, int ldw,
                                               int n0, int k0, int lane) {
    const unsigned short* p =
        W + (size_t)(n0 + (lane & 15)) * ldw + k0 + ((lane >> 4) << 3);
    union { bf16x16 v; bf16x8 h[2]; } u;
    u.h[0] = *(const bf16x8*)(p);
    u.h[1] = *(const bf16x8*)(p + 16);
    return u.v;
}

__device__ __forceinline__ f32x8 wmma_bf16(bf16x16 a, bf16x16 b, f32x8 c) {
    return __builtin_amdgcn_wmma_f32_16x16x32_bf16(
        /*neg_a=*/false, a, /*neg_b=*/false, b,
        /*c_mod=*/(short)0, c, /*reuse_a=*/false, /*reuse_b=*/false);
}

// ---------------------------------------------------------------------------
// Utility kernels
// ---------------------------------------------------------------------------
__global__ void cvt_f32_bf16_kernel(const float* __restrict__ src,
                                    unsigned short* __restrict__ dst, int n) {
    int i = blockIdx.x * blockDim.x + threadIdx.x;
    if (i < n) dst[i] = f2bf(src[i]);
}

__global__ void zero_u32_kernel(unsigned int* __restrict__ dst, int n) {
    int i = blockIdx.x * blockDim.x + threadIdx.x;
    if (i < n) dst[i] = 0u;
}

// emb_bf stored [T, B, E] (row = t*B + b) so each timestep is a contiguous 64xE tile
__global__ void gather_emb_kernel(const int* __restrict__ x,
                                  const float* __restrict__ embed,
                                  unsigned short* __restrict__ emb_bf) {
    int row = blockIdx.x;          // 0..T*B-1, row = t*B + b
    int t = row >> 6;
    int b = row & 63;
    int idx = x[b * Tdim + t];     // x is [B, T]
    const float* src = embed + (size_t)idx * Edim;
    unsigned short* dst = emb_bf + (size_t)row * Edim;
    for (int j = threadIdx.x; j < Edim; j += blockDim.x) dst[j] = f2bf(src[j]);
}

// ---------------------------------------------------------------------------
// Recurrent step kernels (one wave per 16x16 output tile; EXEC always full)
// ---------------------------------------------------------------------------

// sH highway update: Ws = sH@Whh.T + bhh (+ xt@Wih.T + bih at depth 0)
__global__ void __launch_bounds__(32)
step_h_kernel(const unsigned short* __restrict__ sHbf_in,
              const float* __restrict__ sH_in,
              const unsigned short* __restrict__ Whh_bf,
              const float* __restrict__ bhh,
              const unsigned short* __restrict__ emb_bf,
              const unsigned short* __restrict__ Wih_bf,
              const float* __restrict__ bih,
              float* __restrict__ sH_out,
              unsigned short* __restrict__ sHbf_out,
              int t, int i0) {
    int lane = threadIdx.x;
    int nt = blockIdx.x;   // 0..15 : 16-col tile within HH=256
    int mt = blockIdx.y;   // 0..3  : 16-row tile within B=64
    f32x8 chh = {}; f32x8 ctt = {};
    for (int k = 0; k < HHdim; k += 32) {
        bf16x16 a = load_a_frag(sHbf_in, HHdim, mt * 16, k, lane);
        chh = wmma_bf16(a, load_b_frag(Whh_bf, HHdim, nt * 16,         k, lane), chh);
        ctt = wmma_bf16(a, load_b_frag(Whh_bf, HHdim, nt * 16 + HHdim, k, lane), ctt);
    }
    if (i0) {
        const unsigned short* xt = emb_bf + (size_t)(t * Bdim) * Edim;
        for (int k = 0; k < Edim; k += 32) {
            bf16x16 a = load_a_frag(xt, Edim, mt * 16, k, lane);
            chh = wmma_bf16(a, load_b_frag(Wih_bf, Edim, nt * 16,         k, lane), chh);
            ctt = wmma_bf16(a, load_b_frag(Wih_bf, Edim, nt * 16 + HHdim, k, lane), ctt);
        }
    }
    int col = nt * 16 + (lane & 15);
    float b0 = bhh[col], b1 = bhh[col + HHdim];
    if (i0) { b0 += bih[col]; b1 += bih[col + HHdim]; }
    int rbase = mt * 16 + ((lane >> 4) << 3);
#pragma unroll
    for (int r = 0; r < 8; ++r) {
        int row = rbase + r;
        float hh = chh[r] + b0;
        float tt = ctt[r] + b1;
        float sg = 1.0f / (1.0f + __expf(-tt));
        float old = sH_in[row * HHdim + col];
        float v = tanhf(hh) * sg + old * (1.0f - sg);
        sH_out[row * HHdim + col] = v;
        sHbf_out[row * HHdim + col] = f2bf(v);
    }
}

// z = sH_new @ Wup.T + bup   ([64, HN], K=HH)
__global__ void __launch_bounds__(32)
step_z_kernel(const unsigned short* __restrict__ sHbf,
              const unsigned short* __restrict__ Wup_bf,
              const float* __restrict__ bup,
              float* __restrict__ z) {
    int lane = threadIdx.x;
    int nt = blockIdx.x;   // 0..63
    int mt = blockIdx.y;   // 0..3
    f32x8 c = {};
    for (int k = 0; k < HHdim; k += 32) {
        bf16x16 a = load_a_frag(sHbf, HHdim, mt * 16, k, lane);
        c = wmma_bf16(a, load_b_frag(Wup_bf, HHdim, nt * 16, k, lane), c);
    }
    int col = nt * 16 + (lane & 15);
    float bv = bup[col];
    int rbase = mt * 16 + ((lane >> 4) << 3);
#pragma unroll
    for (int r = 0; r < 8; ++r) z[(rbase + r) * HNdim + col] = c[r] + bv;
}

// sN hyper-highway update: Ws = (sN@Wh.T + bh [+ xt@Wi.T + bi]) * zz
// zz = concat(z, z) -> both halves scale by z[n]; fuses scale + gate.
__global__ void __launch_bounds__(32)
step_n_kernel(const unsigned short* __restrict__ sNbf_in,
              const float* __restrict__ sN_in,
              const unsigned short* __restrict__ Wh_bf,
              const float* __restrict__ bh,
              const unsigned short* __restrict__ emb_bf,
              const unsigned short* __restrict__ Wi_bf,
              const float* __restrict__ bi,
              const float* __restrict__ z,
              float* __restrict__ sN_out,
              unsigned short* __restrict__ sNbf_out,
              unsigned short* __restrict__ h_bf,
              int t, int i0) {
    int lane = threadIdx.x;
    int nt = blockIdx.x;   // 0..63 : cols < HN
    int mt = blockIdx.y;   // 0..3
    f32x8 chh = {}; f32x8 ctt = {};
    for (int k = 0; k < HNdim; k += 32) {
        bf16x16 a = load_a_frag(sNbf_in, HNdim, mt * 16, k, lane);
        chh = wmma_bf16(a, load_b_frag(Wh_bf, HNdim, nt * 16,         k, lane), chh);
        ctt = wmma_bf16(a, load_b_frag(Wh_bf, HNdim, nt * 16 + HNdim, k, lane), ctt);
    }
    if (i0) {
        const unsigned short* xt = emb_bf + (size_t)(t * Bdim) * Edim;
        for (int k = 0; k < Edim; k += 32) {
            bf16x16 a = load_a_frag(xt, Edim, mt * 16, k, lane);
            chh = wmma_bf16(a, load_b_frag(Wi_bf, Edim, nt * 16,         k, lane), chh);
            ctt = wmma_bf16(a, load_b_frag(Wi_bf, Edim, nt * 16 + HNdim, k, lane), ctt);
        }
    }
    int col = nt * 16 + (lane & 15);
    float b0 = bh[col], b1 = bh[col + HNdim];
    if (i0) { b0 += bi[col]; b1 += bi[col + HNdim]; }
    int rbase = mt * 16 + ((lane >> 4) << 3);
#pragma unroll
    for (int r = 0; r < 8; ++r) {
        int row = rbase + r;
        float zv = z[row * HNdim + col];
        float hh = (chh[r] + b0) * zv;
        float tt = (ctt[r] + b1) * zv;
        float sg = 1.0f / (1.0f + __expf(-tt));
        float old = sN_in[row * HNdim + col];
        float v = tanhf(hh) * sg + old * (1.0f - sg);
        sN_out[row * HNdim + col] = v;
        sNbf_out[row * HNdim + col] = f2bf(v);
        h_bf[((size_t)(t * Bdim + row)) * HNdim + col] = f2bf(v);
    }
}

// ---------------------------------------------------------------------------
// Output projection: out[b,t,:] = h[t,b,:] @ Wout.T + bout
// Block tile 64(M) x 128(N), 8 waves (4 M-waves x 2 N-waves), wave = 16x64.
// A and B k-chunks double-buffered through LDS. Staging uses the CDNA5
// async Global->LDS engine (ASYNCcnt) when the builtin is available,
// otherwise global_load_b128 + ds_store_b128.
// LDS row stride 40 halves = 80B keeps 16B alignment and spreads banks.
// h_bf rows are m = t*B + b; output row = b*T + t (remap fused into store).
// ---------------------------------------------------------------------------
__global__ void __launch_bounds__(256)
out_gemm_kernel(const unsigned short* __restrict__ h_bf,
                const unsigned short* __restrict__ Wout_bf,
                const float* __restrict__ bout,
                float* __restrict__ out) {
    __shared__ __align__(16) unsigned short Ab[2][BLK_M][LDSW];
    __shared__ __align__(16) unsigned short Bb[2][BLK_N][LDSW];
    const int K = HNdim;

    int tid  = threadIdx.x;
    int lane = tid & 31;
    int wave = tid >> 5;
    int wm = wave >> 1;               // 0..3 : 16-row band
    int wn = wave & 1;                // 0..1 : 64-col band
    int nBlock = blockIdx.x * BLK_N;
    int mBlock = blockIdx.y * BLK_M;

    // staging maps: A chunk 64x32 (16B per thread), B chunk 128x32 (32B per thread)
    int arow = tid >> 2, aq = tid & 3;
    int brow = tid >> 1, bh2 = tid & 1;
    const unsigned short* Ag = h_bf    + (size_t)(mBlock + arow) * K + aq * 8;
    const unsigned short* Bg = Wout_bf + (size_t)(nBlock + brow) * K + bh2 * 16;

    auto stage = [&](int k0, int buf) {
#if HAVE_ASYNC_LDS
        __builtin_amdgcn_global_load_async_to_lds_b128(
            as_global_b128(Ag + k0),     as_lds_b128(&Ab[buf][arow][aq * 8]),       0, 0);
        __builtin_amdgcn_global_load_async_to_lds_b128(
            as_global_b128(Bg + k0),     as_lds_b128(&Bb[buf][brow][bh2 * 16]),     0, 0);
        __builtin_amdgcn_global_load_async_to_lds_b128(
            as_global_b128(Bg + k0 + 8), as_lds_b128(&Bb[buf][brow][bh2 * 16 + 8]), 0, 0);
#else
        bf16x8 av  = *(const bf16x8*)(Ag + k0);
        bf16x8 bv0 = *(const bf16x8*)(Bg + k0);
        bf16x8 bv1 = *(const bf16x8*)(Bg + k0 + 8);
        *(bf16x8*)&Ab[buf][arow][aq * 8]       = av;
        *(bf16x8*)&Bb[buf][brow][bh2 * 16]     = bv0;
        *(bf16x8*)&Bb[buf][brow][bh2 * 16 + 8] = bv1;
#endif
    };

    int aRowL = wm * 16 + (lane & 15);
    int aOff  = (lane >> 4) << 3;     // 0 or 8
    int bOff  = (lane >> 4) << 3;     // 0 or 8

    f32x8 acc[4] = {{}, {}, {}, {}};

    stage(0, 0);
#if HAVE_ASYNC_LDS
    asm volatile("s_wait_asynccnt 0" ::: "memory");
#endif
    __syncthreads();

    const int nK = K / 32;
    for (int ks = 0; ks < nK; ++ks) {
        int buf = ks & 1;
        if (ks + 1 < nK) stage((ks + 1) * 32, buf ^ 1);
#if !HAVE_ASYNC_LDS
        if (ks + 2 < nK) {            // global_prefetch_b8 two chunks ahead
            __builtin_prefetch(Ag + (ks + 2) * 32, 0, 0);
            __builtin_prefetch(Bg + (ks + 2) * 32, 0, 0);
        }
#endif
        union { bf16x16 v; bf16x8 h[2]; } ua;
        ua.h[0] = *(const bf16x8*)&Ab[buf][aRowL][aOff];
        ua.h[1] = *(const bf16x8*)&Ab[buf][aRowL][aOff + 16];
        bf16x16 a = ua.v;
#pragma unroll
        for (int j = 0; j < 4; ++j) {
            int brL = wn * 64 + j * 16 + (lane & 15);
            union { bf16x16 v; bf16x8 h[2]; } ub;
            ub.h[0] = *(const bf16x8*)&Bb[buf][brL][bOff];
            ub.h[1] = *(const bf16x8*)&Bb[buf][brL][bOff + 16];
            acc[j] = wmma_bf16(a, ub.v, acc[j]);
        }
#if HAVE_ASYNC_LDS
        asm volatile("s_wait_asynccnt 0" ::: "memory");
#endif
        __syncthreads();
    }

    int rbase = mBlock + wm * 16 + ((lane >> 4) << 3);
#pragma unroll
    for (int j = 0; j < 4; ++j) {
        int col = nBlock + wn * 64 + j * 16 + (lane & 15);
        if (col < Vdim) {
            float bv = bout[col];
            f32x8 c = acc[j];
#pragma unroll
            for (int r = 0; r < 8; ++r) {
                int m = rbase + r;                       // m = t*B + b
                int orow = (m & 63) * Tdim + (m >> 6);   // b*T + t
                out[(size_t)orow * Vdim + col] = c[r] + bv;
            }
        }
    }
}

// ---------------------------------------------------------------------------
// Host orchestration
// ---------------------------------------------------------------------------
extern "C" void kernel_launch(void* const* d_in, const int* in_sizes, int n_in,
                              void* d_out, int out_size, void* d_ws, size_t ws_size,
                              hipStream_t stream) {
    const int*   x     = (const int*)  d_in[0];
    const float* embed = (const float*)d_in[1];
    const float* Whh   = (const float*)d_in[2];
    const float* bhh   = (const float*)d_in[3];
    const float* Wih   = (const float*)d_in[4];
    const float* bih   = (const float*)d_in[5];
    const float* Wh    = (const float*)d_in[6];
    const float* bh    = (const float*)d_in[7];
    const float* Wi    = (const float*)d_in[8];
    const float* bi    = (const float*)d_in[9];
    const float* Wup   = (const float*)d_in[10];
    const float* bup   = (const float*)d_in[11];
    const float* Wout  = (const float*)d_in[12];
    const float* bout  = (const float*)d_in[13];
    float* out = (float*)d_out;

    // workspace carve-up (256B aligned)
    char* p = (char*)d_ws;
    auto alloc = [&](size_t bytes) -> void* {
        void* r = (void*)p;
        p += (bytes + 255) & ~(size_t)255;
        return r;
    };
    unsigned short* emb_bf  = (unsigned short*)alloc((size_t)Tdim * Bdim * Edim * 2);
    unsigned short* Whh_bf  = (unsigned short*)alloc((size_t)2 * HHdim * HHdim * 2);
    unsigned short* Wih_bf  = (unsigned short*)alloc((size_t)2 * HHdim * Edim * 2);
    unsigned short* Wh_bf   = (unsigned short*)alloc((size_t)2 * HNdim * HNdim * 2);
    unsigned short* Wi_bf   = (unsigned short*)alloc((size_t)2 * HNdim * Edim * 2);
    unsigned short* Wup_bf  = (unsigned short*)alloc((size_t)HNdim * HHdim * 2);
    unsigned short* Wout_bf = (unsigned short*)alloc((size_t)NPAD * HNdim * 2); // padded rows
    float*          sH[2]; unsigned short* sHbf[2];
    float*          sN[2]; unsigned short* sNbf[2];
    sH[0]   = (float*)alloc((size_t)Bdim * HHdim * 4);
    sH[1]   = (float*)alloc((size_t)Bdim * HHdim * 4);
    sHbf[0] = (unsigned short*)alloc((size_t)Bdim * HHdim * 2);
    sHbf[1] = (unsigned short*)alloc((size_t)Bdim * HHdim * 2);
    sN[0]   = (float*)alloc((size_t)Bdim * HNdim * 4);
    sN[1]   = (float*)alloc((size_t)Bdim * HNdim * 4);
    sNbf[0] = (unsigned short*)alloc((size_t)Bdim * HNdim * 2);
    sNbf[1] = (unsigned short*)alloc((size_t)Bdim * HNdim * 2);
    float*          zbuf = (float*)alloc((size_t)Bdim * HNdim * 4);
    unsigned short* h_bf = (unsigned short*)alloc((size_t)Tdim * Bdim * HNdim * 2);

    // 1) weight conversion f32 -> bf16
    struct { const float* s; unsigned short* d; int n; } cvts[] = {
        { Whh,  Whh_bf,  2 * HHdim * HHdim },
        { Wih,  Wih_bf,  2 * HHdim * Edim  },
        { Wh,   Wh_bf,   2 * HNdim * HNdim },
        { Wi,   Wi_bf,   2 * HNdim * Edim  },
        { Wup,  Wup_bf,  HNdim * HHdim     },
        { Wout, Wout_bf, Vdim * HNdim      },
    };
    for (auto& c : cvts)
        cvt_f32_bf16_kernel<<<(c.n + 255) / 256, 256, 0, stream>>>(c.s, c.d, c.n);

    // 2) embedding gather ([T, B, E], bf16)
    gather_emb_kernel<<<Tdim * Bdim, 256, 0, stream>>>(x, embed, emb_bf);

    // 3) zero initial carries (buffer 0)
    zero_u32_kernel<<<(Bdim * HHdim + 255) / 256, 256, 0, stream>>>(
        (unsigned int*)sH[0], Bdim * HHdim);
    zero_u32_kernel<<<(Bdim * HHdim / 2 + 255) / 256, 256, 0, stream>>>(
        (unsigned int*)sHbf[0], Bdim * HHdim / 2);
    zero_u32_kernel<<<(Bdim * HNdim + 255) / 256, 256, 0, stream>>>(
        (unsigned int*)sN[0], Bdim * HNdim);
    zero_u32_kernel<<<(Bdim * HNdim / 2 + 255) / 256, 256, 0, stream>>>(
        (unsigned int*)sNbf[0], Bdim * HNdim / 2);

    // 4) recurrence: T timesteps x D depth iterations (double-buffered carries)
    int cur = 0;
    for (int t = 0; t < Tdim; ++t) {
        for (int i = 0; i < Ddepth; ++i) {
            int i0 = (i == 0) ? 1 : 0;
            int nxt = cur ^ 1;
            step_h_kernel<<<dim3(HHdim / 16, Bdim / 16), 32, 0, stream>>>(
                sHbf[cur], sH[cur], Whh_bf, bhh, emb_bf, Wih_bf, bih,
                sH[nxt], sHbf[nxt], t, i0);
            step_z_kernel<<<dim3(HNdim / 16, Bdim / 16), 32, 0, stream>>>(
                sHbf[nxt], Wup_bf, bup, zbuf);
            step_n_kernel<<<dim3(HNdim / 16, Bdim / 16), 32, 0, stream>>>(
                sNbf[cur], sN[cur], Wh_bf, bh, emb_bf, Wi_bf, bi, zbuf,
                sN[nxt], sNbf[nxt], h_bf, t, i0);
            cur = nxt;
        }
    }

    // 5) output projection (block-tiled, LDS double-buffered, [T,B]->[B,T] remap)
    out_gemm_kernel<<<dim3(NPAD / BLK_N, (Tdim * Bdim) / BLK_M), 256, 0, stream>>>(
        h_bf, Wout_bf, bout, out);
}
